// RNNFiLM_21328807591981
// MI455X (gfx1250) — matrix operand order repaired and checked by reading, not compiled, not run on hardware
//
#include <hip/hip_runtime.h>
#include <hip/hip_bf16.h>

// ---------------------------------------------------------------------------
// Types for CDNA5 WMMA (wave32): V_WMMA_F32_16X16X32_BF16
// ---------------------------------------------------------------------------
typedef __attribute__((ext_vector_type(16))) __bf16 v16bf;
typedef __attribute__((ext_vector_type(8)))  float  v8f;
typedef __attribute__((ext_vector_type(4)))  int    i32x4;

union ABu { uint4 q[2]; v16bf v; };

__device__ __forceinline__ v8f wmma_bf16(v16bf a, v16bf b, v8f c) {
  // 8 args: (neg_a, A, neg_b, B, c_mod, C, reuse_a, reuse_b)
  return __builtin_amdgcn_wmma_f32_16x16x32_bf16(false, a, false, b, (short)0, c, false, false);
}

__device__ __forceinline__ unsigned short f2bf(float x) {
  __bf16 h = (__bf16)x;
  return __builtin_bit_cast(unsigned short, h);
}
__device__ __forceinline__ float bf2f(unsigned short u) {
  return __builtin_bit_cast(float, ((unsigned)u) << 16);
}
__device__ __forceinline__ float sigm(float x) { return 1.f / (1.f + __expf(-x)); }
__device__ __forceinline__ float tanh_fast(float x) { return 2.f * sigm(2.f * x) - 1.f; }

// Problem constants
#define BB   4
#define CC   2
#define EE   512
#define HH   1024
#define TT   1024
#define FF   513
#define G3   3072     // 3*H
#define NWG  16       // workgroups in persistent recurrence kernel

// Async global->LDS copy (gfx1250). Guarded so compilation never regresses.
#if __has_builtin(__builtin_amdgcn_global_load_async_to_lds_b128)
#define HAVE_ASYNC_LDS 1
#else
#define HAVE_ASYNC_LDS 0
#endif

__device__ __forceinline__ void wait_asynccnt0() {
#if __has_builtin(__builtin_amdgcn_s_wait_asynccnt)
  __builtin_amdgcn_s_wait_asynccnt(0);
#else
  asm volatile("s_wait_asynccnt 0x0" ::: "memory");
#endif
}

// ---------------------------------------------------------------------------
// Pack W (f32 row-major [nrows][K]) into WMMA B-operand bf16 layout:
//   out dword index = ((ntile*KB + kb)*32 + lane)*8 + r
//   holds bf16 pair { W[n, k], W[n, k+1] } with n = n0 + lane%16,
//   k = kb*32 + 8*(lane/16) + 2r + (r>=4 ? 8 : 0)
// ---------------------------------------------------------------------------
__global__ void pack_b_kernel(const float* __restrict__ W, unsigned* __restrict__ out,
                              int nrows, int K, int total) {
  int idx = blockIdx.x * 256 + threadIdx.x;
  if (idx >= total) return;
  int r     = idx & 7;
  int lane  = (idx >> 3) & 31;
  int KB    = K >> 5;
  int kb    = (idx >> 8) % KB;
  int ntile = (idx >> 8) / KB;
  int n = ntile * 16 + (lane & 15);
  int k = kb * 32 + ((lane >> 4) << 3) + 2 * r + ((r >> 2) << 3);
  float lo = (n < nrows) ? W[(size_t)n * K + k]     : 0.f;
  float hi = (n < nrows) ? W[(size_t)n * K + k + 1] : 0.f;
  out[idx] = (unsigned)f2bf(lo) | ((unsigned)f2bf(hi) << 16);
}

// ---------------------------------------------------------------------------
// gx0[b][g] = b_ih0[g] + dot(emb[b,:], w_ih0[g,:])   (constant across time)
// ---------------------------------------------------------------------------
__global__ void gx0_kernel(const float* __restrict__ emb, const float* __restrict__ w,
                           const float* __restrict__ bias, float* __restrict__ out) {
  int idx = blockIdx.x * 256 + threadIdx.x;
  if (idx >= BB * G3) return;
  int b = idx / G3, g = idx % G3;
  float s = bias[g];
  const float* e  = emb + b * EE;
  const float* wr = w + (size_t)g * EE;
  for (int k = 0; k < EE; ++k) s = fmaf(e[k], wr[k], s);
  out[idx] = s;
}

__global__ void zero_kernel(float* __restrict__ p, int n) {
  int i = blockIdx.x * 256 + threadIdx.x;
  if (i < n) p[i] = 0.f;
}

// ---------------------------------------------------------------------------
// Persistent GRU recurrence. 16 WGs x 128 threads (4 waves). Wave owns 16
// hidden columns j0..j0+15; computes gates r/z/n via 3 WMMA tile chains.
// gh = h @ W_hh^T. Grid barrier (agent-scope atomic) between steps.
// h is mirrored in bf16 (h_bf) so per-step LDS staging is a raw byte copy,
// done with GLOBAL_LOAD_ASYNC_TO_LDS_B128 when available (ASYNCcnt path).
// ---------------------------------------------------------------------------
__global__ __launch_bounds__(128, 1) void gru_rec_kernel(
    const unsigned* __restrict__ pk_whh,        // packed [192][32][32][8]
    const float* __restrict__ gx_const,         // [4][3072] or nullptr
    const unsigned short* __restrict__ gx_seq,  // bf16 [4096][3072] or nullptr
    const float* __restrict__ b_hh,             // [3072]
    float* __restrict__ h_buf,                  // [2][4][1024] f32 ping-pong
    unsigned short* __restrict__ h_bf,          // [2][4][1024] bf16 ping-pong
    unsigned short* __restrict__ y_out,         // bf16 [4096][1024]
    unsigned* __restrict__ ctr) {
  __shared__ unsigned short h_lds[16 * HH];     // 32 KB, rows 4..15 zero pad
  const int tid  = threadIdx.x;
  const int lane = tid & 31;
  const int wv   = tid >> 5;
  const int j0   = blockIdx.x * 64 + wv * 16;
  const int tR = j0 >> 4;

  for (int i = tid; i < 12 * HH; i += 128) h_lds[4 * HH + i] = 0;

  const int j = j0 + (lane & 15);
  const float bhr = b_hh[j], bhz = b_hh[HH + j], bhn = b_hh[2 * HH + j];
  float gxr_c[4], gxz_c[4], gxn_c[4];
  if (gx_const) {
    #pragma unroll
    for (int b = 0; b < 4; ++b) {
      gxr_c[b] = gx_const[b * G3 + j];
      gxz_c[b] = gx_const[b * G3 + HH + j];
      gxn_c[b] = gx_const[b * G3 + 2 * HH + j];
    }
  }
  const int base1 = (lane & 16) ? 8 : 0;
  const int m = lane & 15;
  // Loop-invariant packed-B base pointers (uint4 units; stride 64/uint4 per kb)
  const uint4* prB0 = (const uint4*)pk_whh + ((size_t)(tR      ) * 1024 + lane) * 2;
  const uint4* pzB0 = (const uint4*)pk_whh + ((size_t)(tR +  64) * 1024 + lane) * 2;
  const uint4* pnB0 = (const uint4*)pk_whh + ((size_t)(tR + 128) * 1024 + lane) * 2;

  for (int t = 0; t < TT; ++t) {
    const int par = t & 1;
    // ---- stage bf16 h into LDS (8 KB) ----
#if HAVE_ASYNC_LDS
    {
      const char* gsrc = (const char*)(h_bf + par * BB * HH) + tid * 16;
      char*       ldst = (char*)h_lds + tid * 16;
      #pragma unroll
      for (int i = 0; i < 4; ++i)
        __builtin_amdgcn_global_load_async_to_lds_b128(
            (__attribute__((address_space(1))) i32x4*)(gsrc + i * 2048),
            (__attribute__((address_space(3))) i32x4*)(ldst + i * 2048), 0, 0);
      wait_asynccnt0();
    }
#else
    for (int i = tid; i < BB * HH; i += 128)
      h_lds[i] = h_bf[par * BB * HH + i];
#endif
    __syncthreads();

    v8f aR = {}; v8f aZ = {}; v8f aN = {};
    const unsigned short* ap = &h_lds[m * HH + base1];
    const uint4* pr = prB0;
    const uint4* pz = pzB0;
    const uint4* pn = pnB0;
    for (int kb = 0; kb < 32; ++kb) {
      ABu A;
      A.q[0] = *(const uint4*)ap;
      A.q[1] = *(const uint4*)(ap + 16);
      ap += 32;
      ABu Br, Bz, Bn;
      Br.q[0] = pr[0]; Br.q[1] = pr[1]; pr += 64;
      Bz.q[0] = pz[0]; Bz.q[1] = pz[1]; pz += 64;
      Bn.q[0] = pn[0]; Bn.q[1] = pn[1]; pn += 64;
      aR = wmma_bf16(A.v, Br.v, aR);
      aZ = wmma_bf16(A.v, Bz.v, aZ);
      aN = wmma_bf16(A.v, Bn.v, aN);
    }

    if (lane < 16) {  // D layout: lanes 0-15 hold M=0..7 (regs); rows 0-3 are real batches
      #pragma unroll
      for (int b = 0; b < 4; ++b) {
        float ghr = aR[b] + bhr;
        float ghz = aZ[b] + bhz;
        float ghn = aN[b] + bhn;
        float gxr, gxz, gxn;
        if (gx_const) { gxr = gxr_c[b]; gxz = gxz_c[b]; gxn = gxn_c[b]; }
        else {
          const unsigned short* g = &gx_seq[(size_t)(b * TT + t) * G3 + j];
          gxr = bf2f(g[0]); gxz = bf2f(g[HH]); gxn = bf2f(g[2 * HH]);
        }
        float r = sigm(gxr + ghr);
        float z = sigm(gxz + ghz);
        float n = tanh_fast(gxn + r * ghn);
        float hp = h_buf[par * BB * HH + b * HH + j];
        float hn = (1.f - z) * n + z * hp;
        unsigned short hb = f2bf(hn);
        h_buf[(1 - par) * BB * HH + b * HH + j] = hn;
        h_bf[(1 - par) * BB * HH + b * HH + j] = hb;
        y_out[(size_t)(b * TT + t) * HH + j] = hb;
      }
    }
    __threadfence();
    __syncthreads();
    if (tid == 0) {
      __hip_atomic_fetch_add(ctr, 1u, __ATOMIC_ACQ_REL, __HIP_MEMORY_SCOPE_AGENT);
      const unsigned target = (unsigned)(t + 1) * NWG;
      while (__hip_atomic_load(ctr, __ATOMIC_ACQUIRE, __HIP_MEMORY_SCOPE_AGENT) < target)
        __builtin_amdgcn_s_sleep(1);
    }
    __syncthreads();
  }
}

// ---------------------------------------------------------------------------
// GEMM: C[M][Npad] = A(bf16 [M][K]) @ Bpacked + bias.  8 waves/block, each
// wave computes one 16(M) x 64(N) tile via 4 WMMA accumulator chains.
// ---------------------------------------------------------------------------
__global__ __launch_bounds__(256, 2) void gemm_bpacked_kernel(
    const unsigned short* __restrict__ Abf, const unsigned* __restrict__ Bpk,
    const float* __restrict__ bias, int nbias,
    int M, int Npad, int K, void* __restrict__ Cout, int ldc, int out_bf16) {
  const int lane = threadIdx.x & 31;
  const int wid  = blockIdx.x * 8 + (threadIdx.x >> 5);
  const int mtiles = M >> 4;
  const int mt = wid % mtiles;
  const int ng = wid / mtiles;
  if (ng >= (Npad >> 6)) return;
  const int m0 = mt << 4, n0 = ng << 6;
  const int KB = K >> 5;
  const int base1 = (lane & 16) ? 8 : 0;
  const int row = m0 + (lane & 15);
  const int t0 = n0 >> 4;
  v8f acc0 = {}, acc1 = {}, acc2 = {}, acc3 = {};
  const unsigned short* ap = Abf + (size_t)row * K + base1;
  const uint4* b0 = (const uint4*)Bpk + ((size_t)(t0 + 0) * KB * 32 + lane) * 2;
  const uint4* b1 = (const uint4*)Bpk + ((size_t)(t0 + 1) * KB * 32 + lane) * 2;
  const uint4* b2 = (const uint4*)Bpk + ((size_t)(t0 + 2) * KB * 32 + lane) * 2;
  const uint4* b3 = (const uint4*)Bpk + ((size_t)(t0 + 3) * KB * 32 + lane) * 2;
  for (int kb = 0; kb < KB; ++kb) {
    ABu A;
    A.q[0] = *(const uint4*)ap;
    A.q[1] = *(const uint4*)(ap + 16);
    ap += 32;
    __builtin_prefetch(b0 + 64, 0, 0);  // emits global_prefetch_b8 (gfx1250)
    ABu B0, B1, B2, B3;
    B0.q[0] = b0[0]; B0.q[1] = b0[1]; b0 += 64;
    B1.q[0] = b1[0]; B1.q[1] = b1[1]; b1 += 64;
    B2.q[0] = b2[0]; B2.q[1] = b2[1]; b2 += 64;
    B3.q[0] = b3[0]; B3.q[1] = b3[1]; b3 += 64;
    acc0 = wmma_bf16(A.v, B0.v, acc0);
    acc1 = wmma_bf16(A.v, B1.v, acc1);
    acc2 = wmma_bf16(A.v, B2.v, acc2);
    acc3 = wmma_bf16(A.v, B3.v, acc3);
  }
  const int orow = m0 + ((lane >> 4) << 3);  // lanes 16-31 hold M=8..15
  v8f accs[4] = {acc0, acc1, acc2, acc3};
  #pragma unroll
  for (int i = 0; i < 4; ++i) {
    int col = n0 + i * 16 + (lane & 15);
    float bb = (col < nbias) ? bias[col] : 0.f;
    #pragma unroll
    for (int r = 0; r < 8; ++r) {
      float v = accs[i][r] + bb;
      int rr = orow + r;
      if (out_bf16) ((unsigned short*)Cout)[(size_t)rr * ldc + col] = f2bf(v);
      else          ((float*)Cout)[(size_t)rr * ldc + col] = v;
    }
  }
}

// ---------------------------------------------------------------------------
// out[b1,b2,c,f,t] = x[b2,c,f,t]*scale[b1,t,f] + shift[b1,t,f]
// One block per (b1,b2,c,f); threads stride over t (coalesced x/out).
// ---------------------------------------------------------------------------
__global__ void film_kernel(const float* __restrict__ x, const float* __restrict__ scale,
                            const float* __restrict__ shift, float* __restrict__ out) {
  int id = blockIdx.x;
  int f  = id % FF; id /= FF;
  int c  = id % CC; id /= CC;
  int b2 = id % BB;
  int b1 = id / BB;
  const float* xp = x + ((size_t)(b2 * CC + c) * FF + f) * TT;
  float* op = out + ((size_t)(((b1 * BB) + b2) * CC + c) * FF + f) * TT;
  const float* sc = scale + (size_t)b1 * TT * 576 + f;
  const float* sh = shift + (size_t)b1 * TT * 576 + f;
  for (int t = threadIdx.x; t < TT; t += 256)
    op[t] = fmaf(xp[t], sc[(size_t)t * 576], sh[(size_t)t * 576]);
}

// ---------------------------------------------------------------------------
extern "C" void kernel_launch(void* const* d_in, const int* in_sizes, int n_in,
                              void* d_out, int out_size, void* d_ws, size_t ws_size,
                              hipStream_t stream) {
  (void)in_sizes; (void)n_in; (void)out_size; (void)ws_size;
  const float* x       = (const float*)d_in[0];
  const float* emb     = (const float*)d_in[1];
  const float* w_ih0   = (const float*)d_in[2];
  const float* w_hh0   = (const float*)d_in[3];
  const float* b_ih0   = (const float*)d_in[4];
  const float* b_hh0   = (const float*)d_in[5];
  const float* w_ih1   = (const float*)d_in[6];
  const float* w_hh1   = (const float*)d_in[7];
  const float* b_ih1   = (const float*)d_in[8];
  const float* b_hh1   = (const float*)d_in[9];
  const float* w_scale = (const float*)d_in[10];
  const float* b_scale = (const float*)d_in[11];
  const float* w_shift = (const float*)d_in[12];
  const float* b_shift = (const float*)d_in[13];
  float* out = (float*)d_out;

  char* ws = (char*)d_ws;
  size_t o = 0;
  auto alloc = [&](size_t b) { size_t r = o; o = (o + b + 255) & ~(size_t)255; return r; };
  unsigned*       pk_whh0 = (unsigned*)(ws + alloc(192u * 32 * 256 * 4));
  unsigned*       pk_whh1 = (unsigned*)(ws + alloc(192u * 32 * 256 * 4));
  unsigned*       pk_wih1 = (unsigned*)(ws + alloc(192u * 32 * 256 * 4));
  unsigned*       pk_wsc  = (unsigned*)(ws + alloc(36u * 32 * 256 * 4));
  unsigned*       pk_wsh  = (unsigned*)(ws + alloc(36u * 32 * 256 * 4));
  float*          gx0     = (float*)(ws + alloc(BB * G3 * 4));
  // h_buf (32KB) + h_bf (16KB) + ctr (256B): contiguous, zeroed together
  float*          h_buf   = (float*)(ws + alloc(2u * BB * HH * 4));
  unsigned short* h_bf    = (unsigned short*)(ws + alloc(2u * BB * HH * 2));
  unsigned*       ctr     = (unsigned*)(ws + alloc(256));
  unsigned short* h1      = (unsigned short*)(ws + alloc((size_t)BB * TT * HH * 2));
  unsigned short* gx1     = (unsigned short*)(ws + alloc((size_t)BB * TT * G3 * 2));
  unsigned short* h2      = (unsigned short*)(ws + alloc((size_t)BB * TT * HH * 2));
  float*          scalef  = (float*)(ws + alloc((size_t)BB * TT * 576 * 4));
  float*          shiftf  = (float*)(ws + alloc((size_t)BB * TT * 576 * 4));
  const int nzero = (2 * BB * HH * 4 + 2 * BB * HH * 2 + 256) / 4;  // floats

  // --- Pack all weight matrices into WMMA-B bf16 layout ---
  int tot3072 = 192 * 32 * 256;
  int totF    = 36 * 32 * 256;
  pack_b_kernel<<<(tot3072 + 255) / 256, 256, 0, stream>>>(w_hh0, pk_whh0, G3, HH, tot3072);
  pack_b_kernel<<<(tot3072 + 255) / 256, 256, 0, stream>>>(w_hh1, pk_whh1, G3, HH, tot3072);
  pack_b_kernel<<<(tot3072 + 255) / 256, 256, 0, stream>>>(w_ih1, pk_wih1, G3, HH, tot3072);
  pack_b_kernel<<<(totF + 255) / 256, 256, 0, stream>>>(w_scale, pk_wsc, FF, HH, totF);
  pack_b_kernel<<<(totF + 255) / 256, 256, 0, stream>>>(w_shift, pk_wsh, FF, HH, totF);

  // --- gx0 (constant across time) ---
  gx0_kernel<<<(BB * G3 + 255) / 256, 256, 0, stream>>>(emb, w_ih0, b_ih0, gx0);

  // --- GRU layer 0 recurrence (h0 = 0, barrier counter = 0) ---
  zero_kernel<<<(nzero + 255) / 256, 256, 0, stream>>>(h_buf, nzero);
  gru_rec_kernel<<<NWG, 128, 0, stream>>>(pk_whh0, gx0, nullptr, b_hh0, h_buf, h_bf, h1, ctr);

  // --- gx1 = h1 @ w_ih1^T + b_ih1 (bf16 out) : M=4096, N=3072, K=1024 ---
  gemm_bpacked_kernel<<<(256 * 48) / 8, 256, 0, stream>>>(
      h1, pk_wih1, b_ih1, G3, BB * TT, G3, HH, gx1, G3, 1);

  // --- GRU layer 1 recurrence ---
  zero_kernel<<<(nzero + 255) / 256, 256, 0, stream>>>(h_buf, nzero);
  gru_rec_kernel<<<NWG, 128, 0, stream>>>(pk_whh1, nullptr, gx1, b_hh1, h_buf, h_bf, h2, ctr);

  // --- FiLM heads: scale/shift = h2 @ W^T + b (f32 out, padded N=576) ---
  gemm_bpacked_kernel<<<(256 * 9) / 8, 256, 0, stream>>>(
      h2, pk_wsc, b_scale, FF, BB * TT, 576, HH, scalef, 576, 0);
  gemm_bpacked_kernel<<<(256 * 9) / 8, 256, 0, stream>>>(
      h2, pk_wsh, b_shift, FF, BB * TT, 576, HH, shiftf, 576, 0);

  // --- Broadcast apply: out[b1,b2,c,f,t] ---
  film_kernel<<<BB * BB * CC * FF, 256, 0, stream>>>(x, scalef, shiftf, out);
}
